// SSAKT_16020228014224
// MI455X (gfx1250) — matrix operand order repaired and checked by reading, not compiled, not run on hardware
//
#include <hip/hip_runtime.h>

// ---------------------------------------------------------------------------
// CDNA5 (gfx1250) transformer block forward: QKV -> causal MHA -> proj+LN ->
// FFN(relu)+LN.  All GEMMs and attention matmuls via v_wmma_f32_16x16x32_bf16.
// GEMM uses double-buffered LDS with async global->LDS copies (ASYNCcnt).
// B=16, S=1024, D=512, H=8, DK=DV=64, DFF=2048.
// ---------------------------------------------------------------------------

typedef __attribute__((ext_vector_type(16))) __bf16 v16bf;
typedef __attribute__((ext_vector_type(8)))  float  v8f;

union BFrag { v16bf v; uint4 q[2]; };
union FAcc  { v8f  v; float f[8]; };

#define WMMA_BF16(A, B, C) \
  __builtin_amdgcn_wmma_f32_16x16x32_bf16(false, (A), false, (B), (short)0, (C), false, false)

// ---- async global->LDS path (gfx1250) -------------------------------------
#if defined(__AMDGCN__) && __has_builtin(__builtin_amdgcn_global_load_async_to_lds_b128)
#define HAS_ASYNC_LDS 1
#else
#define HAS_ASYNC_LDS 0
#endif

__device__ __forceinline__ void async_copy_b128(void* lds, const void* gptr) {
#if HAS_ASYNC_LDS
  typedef int gi4 __attribute__((vector_size(16)));
  __builtin_amdgcn_global_load_async_to_lds_b128(
      (__attribute__((address_space(1))) gi4*)gptr,
      (__attribute__((address_space(3))) gi4*)lds, 0, 0);
#else
  *(uint4*)lds = *(const uint4*)gptr;
#endif
}

#if defined(__AMDGCN__) && __has_builtin(__builtin_amdgcn_s_wait_asynccnt)
#define WAIT_ASYNCCNT(n) __builtin_amdgcn_s_wait_asynccnt(n)
#elif HAS_ASYNC_LDS
#define WAIT_ASYNCCNT(n) asm volatile("s_wait_asynccnt %0" ::"n"(n) : "memory")
#else
#define WAIT_ASYNCCNT(n) ((void)0)
#endif

__device__ __forceinline__ float half16_max(float v) {
  v = fmaxf(v, __shfl_xor(v, 1, 32));
  v = fmaxf(v, __shfl_xor(v, 2, 32));
  v = fmaxf(v, __shfl_xor(v, 4, 32));
  v = fmaxf(v, __shfl_xor(v, 8, 32));
  return v;
}
__device__ __forceinline__ float half16_sum(float v) {
  v += __shfl_xor(v, 1, 32);
  v += __shfl_xor(v, 2, 32);
  v += __shfl_xor(v, 4, 32);
  v += __shfl_xor(v, 8, 32);
  return v;
}

// ---------------------------------------------------------------------------
// f32 -> bf16 conversion
// ---------------------------------------------------------------------------
__global__ void cvt_f32_bf16_kernel(const float* __restrict__ in,
                                    __bf16* __restrict__ out, int n) {
  int i = blockIdx.x * 256 + threadIdx.x;
  if (i < n) out[i] = (__bf16)in[i];
}

// ---------------------------------------------------------------------------
// C[M,N] = A[M,K] @ W[N,K]^T (+bias) (+resid) (relu?) ; A,W bf16, acc f32.
// Tile 128x128x32, 256 threads (8 waves, 2x4), 8 WMMA tiles per wave.
// Double-buffered LDS; next tile prefetched with async global->LDS copies.
// flags: 1 = relu, 2 = scatter bf16 output to [b,h,s,dh] (qkv layout)
// ---------------------------------------------------------------------------
__global__ __launch_bounds__(256)
void gemm_bf16_kernel(const __bf16* __restrict__ A, const __bf16* __restrict__ W,
                      const float* __restrict__ bias, const float* __restrict__ resid,
                      float* __restrict__ outF, __bf16* __restrict__ outB,
                      int M, int N, int K, int flags) {
  __shared__ __align__(16) __bf16 sA[2][128][40];
  __shared__ __align__(16) __bf16 sW[2][128][40];

  const int tid  = threadIdx.x;
  const int lane = tid & 31;
  const int wid  = tid >> 5;
  const int wm   = wid & 1;        // 2 waves over M (64 rows each)
  const int wn   = wid >> 1;       // 4 waves over N (32 cols each)
  const int mbase = blockIdx.y * 128;
  const int nbase = blockIdx.x * 128;

  FAcc acc[4][2];
#pragma unroll
  for (int i = 0; i < 4; ++i)
#pragma unroll
    for (int j = 0; j < 2; ++j)
#pragma unroll
      for (int r = 0; r < 8; ++r) acc[i][j].f[r] = 0.0f;

  const int lr = tid >> 1;           // 0..127  (tile row)
  const int lc = (tid & 1) * 16;     // 0 or 16 (16 bf16 = 32B per thread)
  const __bf16* Arow = A + (size_t)(mbase + lr) * K + lc;
  const __bf16* Wrow = W + (size_t)(nbase + lr) * K + lc;

  const int arow = lane & 15;
  const int kh   = lane >> 4;

  auto issue = [&](int buf, int kb) {
    async_copy_b128(&sA[buf][lr][lc],     Arow + kb);
    async_copy_b128(&sA[buf][lr][lc + 8], Arow + kb + 8);
    async_copy_b128(&sW[buf][lr][lc],     Wrow + kb);
    async_copy_b128(&sW[buf][lr][lc + 8], Wrow + kb + 8);
  };

  const int nk = K >> 5;
  issue(0, 0);
  for (int it = 0; it < nk; ++it) {
    const int p = it & 1;
    if (it + 1 < nk) {
      issue(p ^ 1, (it + 1) * 32);   // prefetch next tile into other buffer
      WAIT_ASYNCCNT(4);              // wait only for current buffer's 4 copies
    } else {
      WAIT_ASYNCCNT(0);
    }
    __syncthreads();                 // current buffer visible to all waves

    BFrag af[4], bf2[2];
#pragma unroll
    for (int i = 0; i < 4; ++i) {
      const __bf16* pp = &sA[p][wm * 64 + i * 16 + arow][0];
      af[i].q[0] = *(const uint4*)(pp + kh * 8);        // K = kh*8 .. +7
      af[i].q[1] = *(const uint4*)(pp + 16 + kh * 8);   // K = 16+kh*8 .. +7
    }
#pragma unroll
    for (int j = 0; j < 2; ++j) {
      const __bf16* pp = &sW[p][wn * 32 + j * 16 + arow][kh * 8];
      bf2[j].q[0] = *(const uint4*)(pp);                // K = kh*8 .. +7
      bf2[j].q[1] = *(const uint4*)(pp + 16);
    }
#pragma unroll
    for (int i = 0; i < 4; ++i)
#pragma unroll
      for (int j = 0; j < 2; ++j)
        acc[i][j].v = WMMA_BF16(af[i].v, bf2[j].v, acc[i][j].v);
    __syncthreads();                 // all reads of buffer p done before reuse
  }

  const bool relu = (flags & 1) != 0;
  const bool qkv  = (flags & 2) != 0;
#pragma unroll
  for (int i = 0; i < 4; ++i) {
#pragma unroll
    for (int j = 0; j < 2; ++j) {
      const int mt = mbase + wm * 64 + i * 16 + kh * 8;   // + r below
      const int nt = nbase + wn * 32 + j * 16 + (lane & 15);
      const float bv = bias ? bias[nt] : 0.0f;
#pragma unroll
      for (int r = 0; r < 8; ++r) {
        const int m = mt + r;
        const size_t idx = (size_t)m * N + nt;
        float v = acc[i][j].f[r] + bv;
        if (resid) v += resid[idx];
        if (relu)  v = fmaxf(v, 0.0f);
        if (outF)  outF[idx] = v;
        if (outB) {
          if (qkv) {  // [b,h,s,dh] layout: b=m>>10, s=m&1023, h=nt>>6, dh=nt&63
            const size_t oi =
                ((((size_t)(m >> 10) * 8 + (nt >> 6)) << 10) + (m & 1023)) * 64 +
                (nt & 63);
            outB[oi] = (__bf16)v;
          } else {
            outB[idx] = (__bf16)v;
          }
        }
      }
    }
  }
}

// ---------------------------------------------------------------------------
// Flash attention, causal with diagonal offset `current`:
//   valid(i,j) : j + current <= i ; fully-masked rows output 0
// Q,K,V bf16 [B*H, S, 64]; output bf16 [B*S, H*64] (row-major for next GEMM).
// One block = 8 waves x 16 query rows = 128 rows; grid (S/128, B*H).
// ---------------------------------------------------------------------------
__global__ __launch_bounds__(256)
void flash_attn_kernel(const __bf16* __restrict__ Q, const __bf16* __restrict__ Kg,
                       const __bf16* __restrict__ Vg, const int* __restrict__ curp,
                       __bf16* __restrict__ Og) {
  __shared__ __align__(16) __bf16 sK[32][72];        // K chunk  [kv][dk]
  __shared__ __align__(16) __bf16 sVt[64][40];       // V chunk transposed [dv][kv]
  __shared__ __align__(16) __bf16 sP[8][16][40];     // per-wave P staging

  const int tid  = threadIdx.x;
  const int lane = tid & 31;
  const int wid  = tid >> 5;
  const int bh   = blockIdx.y;         // b*8 + h
  const int b    = bh >> 3;
  const int h    = bh & 7;
  const int qblk = blockIdx.x;
  const int qbase = qblk * 128 + wid * 16;
  const int current = *curp;
  const float scale = 0.125f;          // 1/sqrt(64)

  const size_t bhoff = (size_t)bh * 1024 * 64;
  const int arow = lane & 15;
  const int kh   = lane >> 4;

  // Q fragments for this wave's 16 rows (dk 0..31 and 32..63)
  BFrag aq[2];
  {
    const __bf16* qp = Q + bhoff + (size_t)(qbase + arow) * 64;
#pragma unroll
    for (int f = 0; f < 2; ++f) {
      aq[f].q[0] = *(const uint4*)(qp + f * 32 + kh * 8);
      aq[f].q[1] = *(const uint4*)(qp + f * 32 + 16 + kh * 8);
    }
  }

  FAcc o[4];
#pragma unroll
  for (int t = 0; t < 4; ++t)
#pragma unroll
    for (int r = 0; r < 8; ++r) o[t].f[r] = 0.0f;
  float mrow[8], lrow[8];
#pragma unroll
  for (int r = 0; r < 8; ++r) { mrow[r] = -3.0e38f; lrow[r] = 0.0f; }

  const int kv_end = qblk * 128 + 128;  // block-uniform (barriers stay uniform)
  for (int kvb = 0; kvb < kv_end; kvb += 32) {
    {  // cooperative load: K chunk + transposed V chunk
      const int r  = tid >> 3;            // kv row 0..31
      const int cs = (tid & 7) * 8;       // dk segment
      *(uint4*)&sK[r][cs] =
          *(const uint4*)(Kg + bhoff + (size_t)(kvb + r) * 64 + cs);
      const __bf16* vp = Vg + bhoff + (size_t)(kvb + r) * 64 + cs;
#pragma unroll
      for (int e = 0; e < 8; ++e) sVt[cs + e][r] = vp[e];
    }
    __syncthreads();

    // scores: 2 kv sub-tiles x 2 dk steps = 4 WMMAs
    FAcc s[2];
#pragma unroll
    for (int n = 0; n < 2; ++n) {
      v8f z = {0.f, 0.f, 0.f, 0.f, 0.f, 0.f, 0.f, 0.f};
      BFrag bk;
      const __bf16* p0 = &sK[n * 16 + arow][kh * 8];
      bk.q[0] = *(const uint4*)(p0);
      bk.q[1] = *(const uint4*)(p0 + 16);
      z = WMMA_BF16(aq[0].v, bk.v, z);
      const __bf16* p1 = &sK[n * 16 + arow][32 + kh * 8];
      bk.q[0] = *(const uint4*)(p1);
      bk.q[1] = *(const uint4*)(p1 + 16);
      s[n].v = WMMA_BF16(aq[1].v, bk.v, z);
    }

    // online softmax (per row; each 16-lane half owns rows r / r+8)
    float alpha[8];
#pragma unroll
    for (int r = 0; r < 8; ++r) {
      const int i  = qbase + r + kh * 8;
      const int j0 = kvb + (lane & 15);
      const bool v0 = (j0 + current) <= i;
      const bool v1 = (j0 + 16 + current) <= i;
      const float x0 = v0 ? s[0].f[r] * scale : -3.0e38f;
      const float x1 = v1 ? s[1].f[r] * scale : -3.0e38f;
      const float cm = half16_max(fmaxf(x0, x1));
      const float mn = fmaxf(mrow[r], cm);
      const float p0 = v0 ? __expf(s[0].f[r] * scale - mn) : 0.0f;
      const float p1 = v1 ? __expf(s[1].f[r] * scale - mn) : 0.0f;
      const float a  = (mrow[r] > -1.0e38f) ? __expf(mrow[r] - mn) : 0.0f;
      mrow[r] = mn;
      lrow[r] = lrow[r] * a + half16_sum(p0 + p1);
      alpha[r] = a;
      sP[wid][r + kh * 8][lane & 15]        = (__bf16)p0;
      sP[wid][r + kh * 8][16 + (lane & 15)] = (__bf16)p1;
    }

    // reload P in A-operand layout, rescale O, accumulate O += P @ V
    BFrag pf;
    {
      const __bf16* pp = &sP[wid][arow][0];
      pf.q[0] = *(const uint4*)(pp + kh * 8);
      pf.q[1] = *(const uint4*)(pp + 16 + kh * 8);
    }
#pragma unroll
    for (int t = 0; t < 4; ++t) {
#pragma unroll
      for (int r = 0; r < 8; ++r) o[t].f[r] *= alpha[r];
      BFrag bv;
      const __bf16* vp = &sVt[t * 16 + arow][kh * 8];
      bv.q[0] = *(const uint4*)(vp);
      bv.q[1] = *(const uint4*)(vp + 16);
      o[t].v = WMMA_BF16(pf.v, bv.v, o[t].v);
    }
    __syncthreads();
  }

  // write O / l  (l==0 => fully masked row => 0, matches reference semantics)
#pragma unroll
  for (int t = 0; t < 4; ++t) {
#pragma unroll
    for (int r = 0; r < 8; ++r) {
      const int m  = qbase + r + kh * 8;
      const int cc = h * 64 + t * 16 + (lane & 15);
      const float inv = lrow[r] > 0.0f ? 1.0f / lrow[r] : 0.0f;
      Og[((size_t)b * 1024 + m) * 512 + cc] = (__bf16)(o[t].f[r] * inv);
    }
  }
}

// ---------------------------------------------------------------------------
// LayerNorm over last dim (512). One block per row. Emits f32 and/or bf16.
// ---------------------------------------------------------------------------
__global__ __launch_bounds__(256)
void layernorm_kernel(const float* __restrict__ in, const float* __restrict__ g,
                      const float* __restrict__ bb, float* __restrict__ outF,
                      __bf16* __restrict__ outB, float eps) {
  __shared__ float ws1[8];
  __shared__ float ws2[8];
  const int row = blockIdx.x;
  const int tid = threadIdx.x;
  const float* p = in + (size_t)row * 512;
  const float x0 = p[tid];
  const float x1 = p[tid + 256];

  float s = x0 + x1;
#pragma unroll
  for (int m = 1; m < 32; m <<= 1) s += __shfl_xor(s, m, 32);
  if ((tid & 31) == 0) ws1[tid >> 5] = s;
  __syncthreads();
  float tot = 0.0f;
#pragma unroll
  for (int i = 0; i < 8; ++i) tot += ws1[i];
  const float mu = tot * (1.0f / 512.0f);

  const float d0 = x0 - mu, d1 = x1 - mu;
  float q = d0 * d0 + d1 * d1;
#pragma unroll
  for (int m = 1; m < 32; m <<= 1) q += __shfl_xor(q, m, 32);
  if ((tid & 31) == 0) ws2[tid >> 5] = q;
  __syncthreads();
  float var = 0.0f;
#pragma unroll
  for (int i = 0; i < 8; ++i) var += ws2[i];
  const float rstd = rsqrtf(var * (1.0f / 512.0f) + eps);

  const float y0 = d0 * rstd * g[tid] + bb[tid];
  const float y1 = d1 * rstd * g[tid + 256] + bb[tid + 256];
  const size_t base = (size_t)row * 512;
  if (outF) { outF[base + tid] = y0; outF[base + tid + 256] = y1; }
  if (outB) { outB[base + tid] = (__bf16)y0; outB[base + tid + 256] = (__bf16)y1; }
}

// ---------------------------------------------------------------------------
// Host launcher
// ---------------------------------------------------------------------------
extern "C" void kernel_launch(void* const* d_in, const int* in_sizes, int n_in,
                              void* d_out, int out_size, void* d_ws, size_t ws_size,
                              hipStream_t stream) {
  (void)in_sizes; (void)n_in; (void)out_size; (void)ws_size;
  const float* x    = (const float*)d_in[0];
  const float* wq   = (const float*)d_in[1];
  const float* bq   = (const float*)d_in[2];
  const float* wk   = (const float*)d_in[3];
  const float* bk   = (const float*)d_in[4];
  const float* wv   = (const float*)d_in[5];
  const float* bv   = (const float*)d_in[6];
  const float* wfc  = (const float*)d_in[7];
  const float* bfc  = (const float*)d_in[8];
  const float* ln1g = (const float*)d_in[9];
  const float* ln1b = (const float*)d_in[10];
  const float* w1f  = (const float*)d_in[11];
  const float* b1f  = (const float*)d_in[12];
  const float* w2f  = (const float*)d_in[13];
  const float* b2f  = (const float*)d_in[14];
  const float* ln2g = (const float*)d_in[15];
  const float* ln2b = (const float*)d_in[16];
  const int*   cur  = (const int*)d_in[17];

  constexpr int BB = 16, S = 1024, D = 512, H = 8, DFF = 2048;
  constexpr int M = BB * S;  // 16384
  (void)H;

  char* wp = (char*)d_ws;
  auto take = [&](size_t bytes) -> char* {
    char* p = wp;
    wp += (bytes + 255) & ~(size_t)255;
    return p;
  };
  __bf16* xb    = (__bf16*)take((size_t)M * D * 2);
  __bf16* wqb   = (__bf16*)take((size_t)D * D * 2);
  __bf16* wkb   = (__bf16*)take((size_t)D * D * 2);
  __bf16* wvb   = (__bf16*)take((size_t)D * D * 2);
  __bf16* wfcb  = (__bf16*)take((size_t)D * D * 2);
  __bf16* w1fb  = (__bf16*)take((size_t)DFF * D * 2);
  __bf16* w2fb  = (__bf16*)take((size_t)D * DFF * 2);
  __bf16* qb    = (__bf16*)take((size_t)M * D * 2);   // [b,h,s,64]
  __bf16* kb    = (__bf16*)take((size_t)M * D * 2);   // [b,h,s,64]
  __bf16* vbuf  = (__bf16*)take((size_t)M * D * 2);   // [b,h,s,64]
  __bf16* attnb = (__bf16*)take((size_t)M * D * 2);   // [b*s, h*64]
  float*  res1  = (float*) take((size_t)M * D * 4);
  float*  ln1f  = (float*) take((size_t)M * D * 4);
  __bf16* ln1bf = (__bf16*)take((size_t)M * D * 2);
  __bf16* hb    = (__bf16*)take((size_t)M * DFF * 2);
  float*  res2  = (float*) take((size_t)M * D * 4);

  auto cvt = [&](const float* in, __bf16* out, int n) {
    cvt_f32_bf16_kernel<<<(n + 255) / 256, 256, 0, stream>>>(in, out, n);
  };
  cvt(x,   xb,   M * D);
  cvt(wq,  wqb,  D * D);
  cvt(wk,  wkb,  D * D);
  cvt(wv,  wvb,  D * D);
  cvt(wfc, wfcb, D * D);
  cvt(w1f, w1fb, DFF * D);
  cvt(w2f, w2fb, D * DFF);

  // QKV projections (bf16 out, scattered to [b,h,s,64])
  gemm_bf16_kernel<<<dim3(D / 128, M / 128), 256, 0, stream>>>(
      xb, wqb, bq, nullptr, nullptr, qb, M, D, D, 2);
  gemm_bf16_kernel<<<dim3(D / 128, M / 128), 256, 0, stream>>>(
      xb, wkb, bk, nullptr, nullptr, kb, M, D, D, 2);
  gemm_bf16_kernel<<<dim3(D / 128, M / 128), 256, 0, stream>>>(
      xb, wvb, bv, nullptr, nullptr, vbuf, M, D, D, 2);

  // causal attention
  flash_attn_kernel<<<dim3(S / 128, 16 * 8), 256, 0, stream>>>(
      qb, kb, vbuf, cur, attnb);

  // output projection + bias + residual(x)
  gemm_bf16_kernel<<<dim3(D / 128, M / 128), 256, 0, stream>>>(
      attnb, wfcb, bfc, x, res1, nullptr, M, D, D, 0);

  // LN1 (eps 1e-5) -> f32 residual copy + bf16 GEMM operand
  layernorm_kernel<<<M, 256, 0, stream>>>(res1, ln1g, ln1b, ln1f, ln1bf, 1e-5f);

  // FFN: relu(ln1 @ w1f^T + b1f) -> hb ; hb @ w2f^T + b2f + ln1 -> res2
  gemm_bf16_kernel<<<dim3(DFF / 128, M / 128), 256, 0, stream>>>(
      ln1bf, w1fb, b1f, nullptr, nullptr, hb, M, DFF, D, 1);
  gemm_bf16_kernel<<<dim3(D / 128, M / 128), 256, 0, stream>>>(
      hb, w2fb, b2f, ln1f, res2, nullptr, M, D, DFF, 0);

  // LN2 (eps 1e-6) -> final f32 output
  layernorm_kernel<<<M, 256, 0, stream>>>(res2, ln2g, ln2b, (float*)d_out,
                                          nullptr, 1e-6f);
}
